// BRepGAT_56418690400711
// MI455X (gfx1250) — matrix operand order repaired and checked by reading, not compiled, not run on hardware
//
#include <hip/hip_runtime.h>
#include <hip/hip_bf16.h>
#include <math.h>

// ---------------------------------------------------------------------------
// GAT on MI455X (gfx1250, wave32). Dense h@W via v_wmma_f32_16x16x32_bf16;
// sparse softmax-aggregation via wave-per-edge atomics (L2-resident features).
// ---------------------------------------------------------------------------

typedef __attribute__((ext_vector_type(16))) __bf16 v16bf;
typedef __attribute__((ext_vector_type(2)))  __bf16 v2bf;
typedef __attribute__((ext_vector_type(8)))  float  v8f;

struct BRaw { uint4 a, b; };   // 32B carrier for a v16bf LDS fragment

#define NEG_SLOPE 0.2f

// Monotone float <-> uint encoding so segment-max can use integer atomicMax.
__device__ __forceinline__ unsigned f2ord(float f) {
  unsigned u = __float_as_uint(f);
  return (u & 0x80000000u) ? ~u : (u | 0x80000000u);
}
__device__ __forceinline__ float ord2f(unsigned u) {
  unsigned v = (u & 0x80000000u) ? (u & 0x7fffffffu) : ~u;
  return __uint_as_float(v);
}

// ---------------------------------------------------------------------------
// Dense GEMM: C[N x Co] = A[N x K] * W[K x Co], fp32 in/out, bf16 WMMA inside.
// Requires K % 32 == 0 (true for this net: 128, 640).
// Block: 256 threads = 8 waves; block tile 128 rows x 64 cols; each wave owns
// a 16x64 C strip = four 16x16 WMMA tiles sharing one A fragment.
// W slabs (32k x 64n) double-buffered in LDS, transposed to [n][k] bf16 with
// rows padded to 80B. Staging: each thread owns a 2(k) x 4(n) sub-tile, so
// every column pair is one packed v_cvt_pk_bf16_f32 + ds_store_b32.
// ALIGNED=true when Co % 64 == 0 (vectorized float4 W loads); the fallback
// clamps columns branch-free (clamped columns are never stored -> exact).
// ---------------------------------------------------------------------------
template <bool ALIGNED>
__global__ __launch_bounds__(256) void gat_gemm_wmma(
    const float* __restrict__ A, const float* __restrict__ W,
    float* __restrict__ C, int N, int K, int Co)
{
  __shared__ __bf16 sBT[2][64][40];     // [buf][n][k], rows padded to 80B

  const int lane = threadIdx.x & 31;
  const int wave = threadIdx.x >> 5;    // 0..7
  const int g    = lane >> 4;           // half-wave group (0/1)
  const int r    = lane & 15;

  const int n0 = blockIdx.x * 64;
  const int m0 = blockIdx.y * 128 + wave * 16;

  const int rowStore = m0 + r;
  const int rowLd    = rowStore < N ? rowStore : N - 1;   // clamped load row
  const float* __restrict__ arow = A + (size_t)rowLd * K;

  // Staging coordinates: 2 consecutive k x 4 consecutive n per thread.
  const int skk = (threadIdx.x >> 4) * 2;   // 0,2,...,30
  const int snn = (threadIdx.x & 15) * 4;   // 0,4,...,60

  // Loop-invariant clamped column indices (fallback path only).
  int cidx[4];
#pragma unroll
  for (int j = 0; j < 4; ++j) cidx[j] = min(n0 + snn + j, Co - 1);

  auto stage = [&](const float* __restrict__ p, __bf16 (*dst)[40]) {
    float w0[4], w1[4];
    if constexpr (ALIGNED) {
      const float4 x0 = *(const float4*)p;          // k = skk,   n..n+3
      const float4 x1 = *(const float4*)(p + Co);   // k = skk+1, n..n+3
      w0[0] = x0.x; w0[1] = x0.y; w0[2] = x0.z; w0[3] = x0.w;
      w1[0] = x1.x; w1[1] = x1.y; w1[2] = x1.z; w1[3] = x1.w;
    } else {
#pragma unroll
      for (int j = 0; j < 4; ++j) w0[j] = p[cidx[j]];
#pragma unroll
      for (int j = 0; j < 4; ++j) w1[j] = p[Co + cidx[j]];
    }
#pragma unroll
    for (int j = 0; j < 4; ++j) {
      v2bf pk;
      pk[0] = (__bf16)w0[j];
      pk[1] = (__bf16)w1[j];
      *(v2bf*)&dst[snn + j][skk] = pk;   // 4B-aligned packed pair store
    }
  };

  const size_t wstep = (size_t)32 * Co;
  const float* wp = W + (size_t)skk * Co + (ALIGNED ? (size_t)(n0 + snn) : (size_t)0);

  stage(wp, sBT[0]);
  __syncthreads();

  v8f acc[4] = {};
  const int nsteps = K >> 5;
  const float* wnext = wp + wstep;

  for (int it = 0; it < nsteps; ++it) {
    const int k0 = it << 5;

    // Stage the next W slab into the other LDS buffer (overlaps the WMMAs).
    if (it + 1 < nsteps) {
      stage(wnext, sBT[(it + 1) & 1]);
      wnext += wstep;
    }

    if (k0 + 64 < K) __builtin_prefetch(arow + k0 + 64, 0, 1);

    // A fragment: two contiguous 8-float K-runs per lane -> 4x global_load_b128.
    const float4* ap0 = (const float4*)(arow + k0 + g * 16);
    const float4* ap1 = (const float4*)(arow + k0 + g * 16 + 8);
    const float4 a0 = ap0[0], a1 = ap0[1];
    const float4 a2 = ap1[0], a3 = ap1[1];
    v16bf afrag;
    afrag[0]  = (__bf16)a0.x; afrag[1]  = (__bf16)a0.y;
    afrag[2]  = (__bf16)a0.z; afrag[3]  = (__bf16)a0.w;
    afrag[4]  = (__bf16)a1.x; afrag[5]  = (__bf16)a1.y;
    afrag[6]  = (__bf16)a1.z; afrag[7]  = (__bf16)a1.w;
    afrag[8]  = (__bf16)a2.x; afrag[9]  = (__bf16)a2.y;
    afrag[10] = (__bf16)a2.z; afrag[11] = (__bf16)a2.w;
    afrag[12] = (__bf16)a3.x; afrag[13] = (__bf16)a3.y;
    afrag[14] = (__bf16)a3.z; afrag[15] = (__bf16)a3.w;

    // Four B fragments (32B contiguous LDS runs -> two ds_load_b128 each),
    // all multiplied against the same A fragment.
    __bf16 (*cur)[40] = sBT[it & 1];
#pragma unroll
    for (int t = 0; t < 4; ++t) {
      const v16bf bt =
          __builtin_bit_cast(v16bf, *(const BRaw*)&cur[16 * t + r][g * 16]);
      acc[t] = __builtin_amdgcn_wmma_f32_16x16x32_bf16(
          false, afrag, false, bt, (short)0, acc[t], false, false);
    }

    __syncthreads();   // staging of next buf done; reads of cur buf done
  }

  // C layout: VGPR v -> row m0 + v + 8*g, col n0 + 16*t + r.
#pragma unroll
  for (int t = 0; t < 4; ++t) {
    const int cc = n0 + 16 * t + r;
    if (cc < Co) {
#pragma unroll
      for (int v = 0; v < 8; ++v) {
        const int rr = m0 + v + 8 * g;
        if (rr < N) C[(size_t)rr * Co + cc] = acc[t][v];
      }
    }
  }
}

// ---------------------------------------------------------------------------
// Per-node attention dots: alphaS[i] = hW[i,:].a_src, alphaD[i] = hW[i,:].a_dst
// One wave32 per node, coalesced feature reads, shuffle reduction.
// ---------------------------------------------------------------------------
__global__ __launch_bounds__(256) void gat_dots(
    const float* __restrict__ hW, const float* __restrict__ a_src,
    const float* __restrict__ a_dst, float* __restrict__ alphaS,
    float* __restrict__ alphaD, int N, int Co)
{
  int node = (int)((blockIdx.x * 256 + threadIdx.x) >> 5);
  int lane = threadIdx.x & 31;
  if (node >= N) return;
  const float* row = hW + (size_t)node * Co;
  float s = 0.0f, d = 0.0f;
  for (int c = lane; c < Co; c += 32) {
    float v = row[c];
    s += v * a_src[c];
    d += v * a_dst[c];
  }
#pragma unroll
  for (int off = 16; off > 0; off >>= 1) {
    s += __shfl_down(s, off, 32);
    d += __shfl_down(d, off, 32);
  }
  if (lane == 0) { alphaS[node] = s; alphaD[node] = d; }
}

// Per-node init: segmax = -inf (ordered encoding), den = 0.
__global__ void gat_init_seg(unsigned* __restrict__ segmax,
                             float* __restrict__ den, int N)
{
  int i = blockIdx.x * blockDim.x + threadIdx.x;
  if (i < N) { segmax[i] = f2ord(-INFINITY); den[i] = 0.0f; }
}

// out[i, c] = bias[c]  (aggregation accumulates on top).
__global__ void gat_bias_init(float* __restrict__ out,
                              const float* __restrict__ b, int N, int Co)
{
  size_t i = (size_t)blockIdx.x * blockDim.x + threadIdx.x;
  if (i < (size_t)N * Co) out[i] = b[i % Co];
}

// Edge logits + segment max. Indices >= E are the implicit self-loops.
__global__ void gat_logits(const int* __restrict__ src, const int* __restrict__ dst,
                           const float* __restrict__ aS, const float* __restrict__ aD,
                           float* __restrict__ logit, unsigned* __restrict__ segmax,
                           int E, int N)
{
  int e = blockIdx.x * blockDim.x + threadIdx.x;
  if (e >= E + N) return;
  int s, d;
  if (e < E) { s = src[e]; d = dst[e]; } else { s = d = e - E; }
  float l = aS[s] + aD[d];
  l = (l > 0.0f) ? l : NEG_SLOPE * l;
  logit[e] = l;
  atomicMax(segmax + d, f2ord(l));
}

// exp(l - max) and segment sum.
__global__ void gat_expsum(const int* __restrict__ dst,
                           const float* __restrict__ logit,
                           const unsigned* __restrict__ segmax,
                           float* __restrict__ den, int E, int N)
{
  int e = blockIdx.x * blockDim.x + threadIdx.x;
  if (e >= E + N) return;
  int d = (e < E) ? dst[e] : (e - E);
  float ex = __expf(logit[e] - ord2f(segmax[d]));
  atomicAdd(den + d, ex);
}

// Weighted scatter-add: out[dst,:] += alpha * hW[src,:]. Wave32 per edge;
// hW (51 MB) is L2-resident on MI455X's 192 MB L2 so gathers stay on-chip.
__global__ __launch_bounds__(256) void gat_aggregate(
    const int* __restrict__ src, const int* __restrict__ dst,
    const float* __restrict__ logit, const unsigned* __restrict__ segmax,
    const float* __restrict__ den, const float* __restrict__ hW,
    float* __restrict__ out, int E, int N, int Co)
{
  int edge = (int)((blockIdx.x * 256 + threadIdx.x) >> 5);
  int lane = threadIdx.x & 31;
  if (edge >= E + N) return;
  int s, d;
  if (edge < E) { s = src[edge]; d = dst[edge]; } else { s = d = edge - E; }
  float alpha = __expf(logit[edge] - ord2f(segmax[d])) / den[d];
  const float* hrow = hW + (size_t)s * Co;
  float*       orow = out + (size_t)d * Co;
  for (int c = lane; c < Co; c += 32) {
    atomicAdd(orow + c, alpha * hrow[c]);
  }
}

__global__ void gat_relu(float* __restrict__ x, size_t n)
{
  size_t i = (size_t)blockIdx.x * blockDim.x + threadIdx.x;
  if (i < n) x[i] = fmaxf(x[i], 0.0f);
}

// ---------------------------------------------------------------------------
// Host launcher
// ---------------------------------------------------------------------------
extern "C" void kernel_launch(void* const* d_in, const int* in_sizes, int n_in,
                              void* d_out, int out_size, void* d_ws, size_t ws_size,
                              hipStream_t stream)
{
  (void)n_in; (void)out_size; (void)ws_size;

  const float* x  = (const float*)d_in[0];
  const int*   ei = (const int*)d_in[1];
  const int N = in_sizes[0] / 128;      // 20000
  const int E = in_sizes[1] / 2;        // 320000
  const int* src = ei;
  const int* dst = ei + E;

  const float *Wp[5], *asp[5], *adp[5], *bp[5];
  for (int i = 0; i < 5; ++i) {
    Wp[i]  = (const float*)d_in[2 + 4 * i];
    asp[i] = (const float*)d_in[3 + 4 * i];
    adp[i] = (const float*)d_in[4 + 4 * i];
    bp[i]  = (const float*)d_in[5 + 4 * i];
  }
  const int Ci[5] = {128, 640, 640, 640, 640};
  const int Co[5] = {640, 640, 640, 640, 2};

  // Workspace carve-up (fp32 elements).
  const size_t big = (size_t)N * 640;
  float* f      = (float*)d_ws;
  float* bufA   = f;                    // hW scratch (layers 1-4)
  float* bufB   = f + big;              // ping
  float* bufC   = f + 2 * big;          // pong
  float* alphaS = f + 3 * big;          // N
  float* alphaD = alphaS + N;           // N
  float* den    = alphaD + N;           // N
  unsigned* segmax = (unsigned*)(den + N);          // N
  float* logit  = (float*)(segmax + N);             // E + N
  float* hW5    = logit + (size_t)(E + N);          // N * 2

  // Layer dataflow: h -> hW -> out (out becomes next layer's h).
  const float* hcur[5] = {x, bufB, bufC, bufB, bufC};
  float* hWbuf[5]      = {bufA, bufA, bufA, bufA, hW5};
  float* outbuf[5]     = {bufB, bufC, bufB, bufC, (float*)d_out};

  const int EN = E + N;

  for (int L = 0; L < 5; ++L) {
    const int K = Ci[L], C = Co[L];

    // 1) Dense GEMM via bf16 WMMA.
    dim3 gg((C + 63) / 64, (N + 127) / 128);
    if ((C & 63) == 0) {
      gat_gemm_wmma<true><<<gg, 256, 0, stream>>>(hcur[L], Wp[L], hWbuf[L], N, K, C);
    } else {
      gat_gemm_wmma<false><<<gg, 256, 0, stream>>>(hcur[L], Wp[L], hWbuf[L], N, K, C);
    }

    // 2) Per-node attention dot products (wave per node).
    int dotBlocks = (N * 32 + 255) / 256;
    gat_dots<<<dotBlocks, 256, 0, stream>>>(hWbuf[L], asp[L], adp[L],
                                            alphaS, alphaD, N, C);

    // 3) Init segment state + bias-seed the output accumulator.
    gat_init_seg<<<(N + 255) / 256, 256, 0, stream>>>(segmax, den, N);
    size_t nc = (size_t)N * C;
    gat_bias_init<<<(int)((nc + 255) / 256), 256, 0, stream>>>(outbuf[L], bp[L], N, C);

    // 4) LeakyReLU logits + segment max (edges + implicit self-loops).
    gat_logits<<<(EN + 255) / 256, 256, 0, stream>>>(src, dst, alphaS, alphaD,
                                                     logit, segmax, E, N);

    // 5) exp + segment sum.
    gat_expsum<<<(EN + 255) / 256, 256, 0, stream>>>(dst, logit, segmax, den, E, N);

    // 6) Weighted scatter-add aggregation (wave per edge, L2-resident hW).
    int aggBlocks = (int)(((size_t)EN * 32 + 255) / 256);
    gat_aggregate<<<aggBlocks, 256, 0, stream>>>(src, dst, logit, segmax, den,
                                                 hWbuf[L], outbuf[L], E, N, C);

    // 7) ReLU between layers (not after the last one).
    if (L < 4) {
      gat_relu<<<(int)((nc + 255) / 256), 256, 0, stream>>>(outbuf[L], nc);
    }
  }
}
